// YiJingTransformerLayer_15221364097195
// MI455X (gfx1250) — matrix operand order repaired and checked by reading, not compiled, not run on hardware
//
#include <hip/hip_runtime.h>

// ---------------------------------------------------------------------------
// CDNA5 / gfx1250 implementation of the YiJing transformer layer.
// Matmuls: v_wmma_f32_16x16x32_bf16 (f32 accumulate).
// GEMM tile staging, double-buffered & software-pipelined:
//   A via global_load_async_to_lds_b128 (ASYNCcnt),
//   B via TDM tensor_load_to_lds (TENSORcnt).
// ---------------------------------------------------------------------------

typedef __bf16 bf16;
typedef __attribute__((ext_vector_type(16))) __bf16 v16bf;
typedef __attribute__((ext_vector_type(8)))  __bf16 v8bf;
typedef __attribute__((ext_vector_type(8)))  float  v8f;
typedef __attribute__((ext_vector_type(4)))  unsigned int u32x4;
typedef __attribute__((ext_vector_type(8)))  unsigned int u32x8;

__device__ __forceinline__ v8f wmma_bf16(v16bf a, v16bf b, v8f c) {
  // (neg_a, A, neg_b, B, c_mod, C, reuse_a, reuse_b)
  return __builtin_amdgcn_wmma_f32_16x16x32_bf16(false, a, false, b, (short)0, c,
                                                 false, false);
}

// LDS byte offset (wave-relative) of a __shared__ pointer: generic -> as(3).
__device__ __forceinline__ unsigned lds_off(const void* p) {
  return (unsigned)(unsigned long long)(__attribute__((address_space(3))) const char*)p;
}

// Async global -> LDS, 16 bytes per lane (ASYNCcnt path).
__device__ __forceinline__ void async_ld_b128(unsigned lds_addr, const void* gaddr) {
  asm volatile("global_load_async_to_lds_b128 %0, %1, off"
               :: "v"(lds_addr), "v"((unsigned long long)(size_t)gaddr)
               : "memory");
}
template <int N>
__device__ __forceinline__ void wait_async() {
  asm volatile("s_wait_asynccnt %0" :: "i"(N) : "memory");
}

// TDM: 2-D tile load global -> LDS (TENSORcnt path).  2 descriptor groups.
__device__ __forceinline__ void tdm_load_2d(u32x4 g0, u32x8 g1) {
  asm volatile("tensor_load_to_lds %0, %1" :: "s"(g0), "s"(g1) : "memory");
}

// reduce across the 16 lanes that hold one matrix row of a C-layout tile
__device__ __forceinline__ float rmax16(float x) {
  x = fmaxf(x, __shfl_xor(x, 8, 16));
  x = fmaxf(x, __shfl_xor(x, 4, 16));
  x = fmaxf(x, __shfl_xor(x, 2, 16));
  x = fmaxf(x, __shfl_xor(x, 1, 16));
  return x;
}
__device__ __forceinline__ float rsum16(float x) {
  x += __shfl_xor(x, 8, 16);
  x += __shfl_xor(x, 4, 16);
  x += __shfl_xor(x, 2, 16);
  x += __shfl_xor(x, 1, 16);
  return x;
}

// ---------------------------- constants ------------------------------------
#define BB   2
#define TT   2048
#define CC   1024
#define HH   8
#define DD   128
#define FFN_ 4096
#define ROWS (BB * TT)          // 4096
#define INV_SQRT_D 0.08838834764831845f
#define LN2_10000_OVER64 0.14391156538f   // ln(10000)/64

// ---------------------------------------------------------------------------
// f32 -> bf16 weight conversion
// ---------------------------------------------------------------------------
__global__ __launch_bounds__(256) void cvt_bf16_kernel(const float* __restrict__ in,
                                                       bf16* __restrict__ out, int n) {
  int i = blockIdx.x * 256 + threadIdx.x;
  if (i < n) out[i] = (bf16)in[i];
}

// ---------------------------------------------------------------------------
// LayerNorm (f32 in) -> bf16 out.  One block per row, C = 1024.
// ---------------------------------------------------------------------------
__global__ __launch_bounds__(256) void ln_bf16_kernel(const float* __restrict__ x,
                                                      const float* __restrict__ g,
                                                      const float* __restrict__ b,
                                                      bf16* __restrict__ out) {
  int row = blockIdx.x, tid = threadIdx.x;
  const float* xr = x + (size_t)row * CC;
  float xv[4], s = 0.f, ss = 0.f;
#pragma unroll
  for (int it = 0; it < 4; ++it) {
    float v = xr[tid + 256 * it];
    xv[it] = v; s += v; ss += v * v;
  }
  for (int m = 16; m >= 1; m >>= 1) { s += __shfl_xor(s, m, 32); ss += __shfl_xor(ss, m, 32); }
  __shared__ float shs[8], shss[8], stats[2];
  if ((tid & 31) == 0) { shs[tid >> 5] = s; shss[tid >> 5] = ss; }
  __syncthreads();
  if (tid == 0) {
    float S = 0.f, SS = 0.f;
    for (int i = 0; i < 8; ++i) { S += shs[i]; SS += shss[i]; }
    float mean = S / CC, var = SS / CC - mean * mean;
    stats[0] = mean; stats[1] = rsqrtf(var + 1e-5f);
  }
  __syncthreads();
  float mean = stats[0], rstd = stats[1];
#pragma unroll
  for (int it = 0; it < 4; ++it) {
    int c = tid + 256 * it;
    out[(size_t)row * CC + c] = (bf16)((xv[it] - mean) * rstd * g[c] + b[c]);
  }
}

// ---------------------------------------------------------------------------
// Tiled WMMA GEMM:  out[M,N] = A[M,K] @ W[N,K]^T   (bf16 in, f32 accumulate)
// 256 threads / 8 waves; block tile 128(M) x 64(N); K pipelined in 32-chunks
// with double-buffered LDS: stage(next) overlaps compute(current).
// EPI: 0 = f32 store, 1 = +bias, exact GELU, bf16 store,
//      2 = +resid, f32 store, 3 = +bias +resid, f32 store
// ---------------------------------------------------------------------------
template <int EPI>
__global__ __launch_bounds__(256) void gemm_bf16_wmma(
    const bf16* __restrict__ A, const bf16* __restrict__ W, int M, int N, int K,
    float* __restrict__ outF, bf16* __restrict__ outB,
    const float* __restrict__ bias, const float* __restrict__ resid) {
  __shared__ __align__(16) bf16 As[2][128 * 32];   // 2 x 8KB
  __shared__ __align__(16) bf16 Bs[2][64 * 32];    // 2 x 4KB
  const int tid = threadIdx.x;
  const int lane = tid & 31;
  const int w = tid >> 5;
  const int ln15 = lane & 15;
  const int hi = (lane & 16) ? 1 : 0;
  const int n0 = blockIdx.x * 64;
  const int m0b = blockIdx.y * 128;
  const int m0 = m0b + w * 16;

  v8f acc[4];
#pragma unroll
  for (int nt = 0; nt < 4; ++nt) acc[nt] = v8f{0.f, 0.f, 0.f, 0.f, 0.f, 0.f, 0.f, 0.f};

  // async A staging: 512 x 16B chunks / buffer; thread moves chunk tid, tid+256
  const unsigned aLds0 = lds_off(&As[0][(tid >> 2) * 32 + (tid & 3) * 8]);
  const unsigned aLds1 = aLds0 + 64 * 32 * 2;         // 64 rows further down
  const unsigned aBufStride = 128 * 32 * 2;           // bytes per A buffer
  const bf16* aG0 = A + (size_t)(m0b + (tid >> 2)) * K + (tid & 3) * 8;
  const bf16* aG1 = aG0 + (size_t)64 * K;
  // TDM descriptor invariants (wave-uniform scalars)
  const unsigned bsBase = lds_off(&Bs[0][0]);         // +4096 for buffer 1
  const unsigned long long wTileBase =
      (unsigned long long)(size_t)(W + (size_t)n0 * K);

  // stage one 32-K slice into buffer `buf`
  auto stage = [&](int buf, int kc) {
    async_ld_b128(aLds0 + buf * aBufStride, aG0 + kc);
    async_ld_b128(aLds1 + buf * aBufStride, aG1 + kc);
    if (w == 0) {
      unsigned long long ga = wTileBase + (unsigned)kc * 2u;  // byte address
      u32x4 g0;
      g0.x = 1u;                                   // count=1, user descriptor
      g0.y = bsBase + (unsigned)buf * 4096u;       // lds_addr (bytes)
      g0.z = (unsigned)ga;                         // global_addr[31:0]
      g0.w = ((unsigned)(ga >> 32) & 0x01FFFFFFu)  // global_addr[56:32]
             | (2u << 30);                         // type = 2 (image)
      u32x8 g1;
      g1[0] = (1u << 16);                          // data_size = 1 (2 bytes)
      g1[1] = ((unsigned)K & 0xFFFFu) << 16;       // tensor_dim0 lo16 @bit48
      g1[2] = ((unsigned)K >> 16)                  // tensor_dim0 hi16
              | (((unsigned)N & 0xFFFFu) << 16);   // tensor_dim1 lo16 @bit80
      g1[3] = ((unsigned)N >> 16)                  // tensor_dim1 hi16
              | (32u << 16);                       // tile_dim0 = 32 @bit112
      g1[4] = 64u;                                 // tile_dim1 = 64, tile_dim2 = 0
      g1[5] = (unsigned)K;                         // tensor_dim0_stride lo32
      g1[6] = 0u;                                  // stride hi16, dim1_stride lo16
      g1[7] = 0u;
      tdm_load_2d(g0, g1);
    }
  };
  // 4 WMMAs from buffer `buf`
  auto compute = [&](int buf) {
    const bf16* aOp = &As[buf][(w * 16 + ln15) * 32 + (hi ? 16 : 0)];
    v16bf a;
    ((v8bf*)&a)[0] = *(const v8bf*)aOp;
    ((v8bf*)&a)[1] = *(const v8bf*)(aOp + 8);
#pragma unroll
    for (int nt = 0; nt < 4; ++nt) {
      v16bf bb;
      const bf16* bp = &Bs[buf][(nt * 16 + ln15) * 32 + (hi ? 16 : 0)];
      ((v8bf*)&bb)[0] = *(const v8bf*)bp;
      ((v8bf*)&bb)[1] = *(const v8bf*)(bp + 8);
      acc[nt] = wmma_bf16(a, bb, acc[nt]);
    }
  };

  stage(0, 0);                           // prologue
  for (int kc = 0; kc < K; kc += 64) {   // K % 64 == 0 for all call sites
    // ---- stage buf1(kc+32) while buf0(kc) is consumed ----
    stage(1, kc + 32);
    wait_async<2>();                     // oldest stage (buf0) landed
    if (w == 0) __builtin_amdgcn_s_wait_tensorcnt(1);
    __syncthreads();
    compute(0);
    __syncthreads();
    // ---- stage buf0(kc+64) while buf1(kc+32) is consumed ----
    if (kc + 64 < K) {
      stage(0, kc + 64);
      wait_async<2>();
      if (w == 0) __builtin_amdgcn_s_wait_tensorcnt(1);
    } else {
      wait_async<0>();
      if (w == 0) __builtin_amdgcn_s_wait_tensorcnt(0);
    }
    __syncthreads();
    compute(1);
    __syncthreads();
  }

#pragma unroll
  for (int nt = 0; nt < 4; ++nt) {
#pragma unroll
    for (int j = 0; j < 8; ++j) {
      int row = m0 + j + (hi ? 8 : 0);
      int col = n0 + nt * 16 + ln15;
      size_t o = (size_t)row * N + col;
      float v = acc[nt][j];
      if (EPI == 0) {
        outF[o] = v;
      } else if (EPI == 1) {
        v += bias[col];
        v = 0.5f * v * (1.0f + erff(v * 0.7071067811865475f));
        outB[o] = (bf16)v;
      } else if (EPI == 2) {
        outF[o] = v + resid[o];
      } else {
        outF[o] = v + bias[col] + resid[o];
      }
    }
  }
}

// ---------------------------------------------------------------------------
// Split qkv[B*T, 3C] f32 -> roped q (scaled by 1/sqrt(D)), roped k  [B,H,T,D] bf16
// and v transposed  [B,H,D,T] bf16.  One thread per (b,h,t,d<64) rotate pair.
// ---------------------------------------------------------------------------
__global__ __launch_bounds__(256) void rope_split_kernel(const float* __restrict__ qkv,
                                                         bf16* __restrict__ q,
                                                         bf16* __restrict__ k,
                                                         bf16* __restrict__ vt) {
  int idx = blockIdx.x * 256 + threadIdx.x;  // B*H*T*64
  int i  = idx & 63;
  int t  = (idx >> 6) & (TT - 1);
  int bh = idx >> 17;                        // 64*2048 = 2^17
  int h  = bh & (HH - 1);
  int b  = bh >> 3;
  const float* row = qkv + (size_t)(b * TT + t) * (3 * CC);
  int co = h * DD + i;
  float inv = __expf(-(float)i * LN2_10000_OVER64);
  float ang = (float)t * inv, s, c;
  __sincosf(ang, &s, &c);
  float q0 = row[co],            q1 = row[co + 64];
  float k0 = row[CC + co],       k1 = row[CC + co + 64];
  float v0 = row[2 * CC + co],   v1 = row[2 * CC + co + 64];
  float qr0 = q0 * c - q1 * s, qr1 = q1 * c + q0 * s;
  float kr0 = k0 * c - k1 * s, kr1 = k1 * c + k0 * s;
  size_t qo = ((size_t)bh * TT + t) * DD;
  q[qo + i]       = (bf16)(qr0 * INV_SQRT_D);
  q[qo + i + 64]  = (bf16)(qr1 * INV_SQRT_D);
  k[qo + i]       = (bf16)kr0;
  k[qo + i + 64]  = (bf16)kr1;
  size_t vo = ((size_t)bh * DD + i) * TT + t;
  vt[vo]              = (bf16)v0;
  vt[vo + 64 * TT]    = (bf16)v1;
}

// ---------------------------------------------------------------------------
// qp/kp: trigram projections of roped q/k (first 3 dims, unscaled, f32 exact)
// ---------------------------------------------------------------------------
__global__ __launch_bounds__(256) void qpkp_kernel(const float* __restrict__ qkv,
                                                   float* __restrict__ qp,
                                                   float* __restrict__ kp) {
  int idx = blockIdx.x * 256 + threadIdx.x;  // B*H*T = 32768
  if (idx >= BB * HH * TT) return;
  int t  = idx & (TT - 1);
  int bh = idx >> 11;
  int h  = bh & (HH - 1);
  int b  = bh >> 3;
  const float* row = qkv + (size_t)(b * TT + t) * (3 * CC);
  float accq = 0.f, acck = 0.f;
#pragma unroll
  for (int j = 0; j < 3; ++j) {
    int co = h * DD + j;
    float inv = __expf(-(float)j * LN2_10000_OVER64);
    float ang = (float)t * inv, s, c;
    __sincosf(ang, &s, &c);
    float qr = row[co] * c - row[co + 64] * s;
    float kr = row[CC + co] * c - row[CC + co + 64] * s;
    float dj = (((h >> j) & 1) ? 1.f : -1.f) * 0.57735026919f;
    accq += qr * dj; acck += kr * dj;
  }
  qp[idx] = accq; kp[idx] = acck;
}

// ---------------------------------------------------------------------------
// Flash attention, causal, with fused rank-1 trigram geo path.
// One wave per 16-query tile; keys swept in 32-wide tiles via WMMA.
// ---------------------------------------------------------------------------
__global__ __launch_bounds__(256) void flash_kernel(
    const bf16* __restrict__ q, const bf16* __restrict__ k,
    const bf16* __restrict__ vt, const float* __restrict__ qp,
    const float* __restrict__ kp, const float* __restrict__ head_scales,
    bf16* __restrict__ attn_out) {
  __shared__ __align__(16) bf16 pT[8][16 * 32];
  __shared__ __align__(16) bf16 p2T[8][16 * 32];

  const int lane = threadIdx.x & 31;
  const int w = threadIdx.x >> 5;
  const int gw = blockIdx.x * 8 + w;
  const int qt = gw & (TT / 16 - 1);   // 128 query tiles
  const int bh = gw >> 7;
  const int h = bh & (HH - 1);
  const int bidx = bh >> 3;
  const int qbase = qt * 16;
  const int ln15 = lane & 15;
  const int hi = (lane & 16) ? 1 : 0;
  const int half8 = hi ? 8 : 0;

  const bf16* qb = q + (size_t)bh * TT * DD;
  const bf16* kb = k + (size_t)bh * TT * DD;
  const bf16* vb = vt + (size_t)bh * DD * TT;
  const float* qpb = qp + (size_t)bh * TT;
  const float* kpb = kp + (size_t)bh * TT;

  v16bf qA[4];
  {
    const bf16* base = qb + (size_t)(qbase + ln15) * DD + (hi ? 16 : 0);
#pragma unroll
    for (int dc = 0; dc < 4; ++dc) {
      ((v8bf*)&qA[dc])[0] = *(const v8bf*)(base + dc * 32);
      ((v8bf*)&qA[dc])[1] = *(const v8bf*)(base + dc * 32 + 8);
    }
  }
  float qpr[8];
#pragma unroll
  for (int j = 0; j < 8; ++j) qpr[j] = qpb[qbase + j + half8];
  const float hs = head_scales[h];

  v8f O[8], O2[8];
#pragma unroll
  for (int c = 0; c < 8; ++c) {
    O[c]  = v8f{0.f, 0.f, 0.f, 0.f, 0.f, 0.f, 0.f, 0.f};
    O2[c] = v8f{0.f, 0.f, 0.f, 0.f, 0.f, 0.f, 0.f, 0.f};
  }
  float mrow[8], l[8], m2row[8], l2[8];
#pragma unroll
  for (int j = 0; j < 8; ++j) { mrow[j] = -3.0e38f; l[j] = 0.f; m2row[j] = -3.0e38f; l2[j] = 0.f; }

  bf16* pw = pT[w];
  bf16* p2w = p2T[w];

  const int nTiles = qbase / 32 + 1;
  for (int ti = 0; ti < nTiles; ++ti) {
    const int sb = ti * 32;
    v8f S0 = v8f{0.f, 0.f, 0.f, 0.f, 0.f, 0.f, 0.f, 0.f};
    v8f S1 = S0;
#pragma unroll
    for (int dc = 0; dc < 4; ++dc) {
      v16bf b0, b1;
      const bf16* kp0 = kb + (size_t)(sb + ln15) * DD + dc * 32 + (hi ? 16 : 0);
      ((v8bf*)&b0)[0] = *(const v8bf*)kp0;
      ((v8bf*)&b0)[1] = *(const v8bf*)(kp0 + 8);
      const bf16* kp1 = kp0 + 16 * DD;
      ((v8bf*)&b1)[0] = *(const v8bf*)kp1;
      ((v8bf*)&b1)[1] = *(const v8bf*)(kp1 + 8);
      S0 = wmma_bf16(qA[dc], b0, S0);
      S1 = wmma_bf16(qA[dc], b1, S1);
    }
    const bool needMask = (sb + 31 > qbase);
    const float kp0v = kpb[sb + ln15];
    const float kp1v = kpb[sb + 16 + ln15];

#pragma unroll
    for (int j = 0; j < 8; ++j) {
      const int row = qbase + j + half8;
      float s0 = S0[j], s1 = S1[j];
      float g0 = qpr[j] * kp0v, g1 = qpr[j] * kp1v;
      if (needMask) {
        if (sb + ln15 > row)      { s0 = -3.0e38f; g0 = -3.0e38f; }
        if (sb + 16 + ln15 > row) { s1 = -3.0e38f; g1 = -3.0e38f; }
      }
      {
        float mn = fmaxf(mrow[j], rmax16(fmaxf(s0, s1)));
        float corr = __expf(mrow[j] - mn);
        mrow[j] = mn;
        float p0 = __expf(s0 - mn), p1 = __expf(s1 - mn);
        l[j] = l[j] * corr + rsum16(p0 + p1);
#pragma unroll
        for (int c = 0; c < 8; ++c) O[c][j] *= corr;
        pw[(j + half8) * 32 + ln15]      = (bf16)p0;
        pw[(j + half8) * 32 + 16 + ln15] = (bf16)p1;
      }
      {
        float mn = fmaxf(m2row[j], rmax16(fmaxf(g0, g1)));
        float corr = __expf(m2row[j] - mn);
        m2row[j] = mn;
        float p0 = __expf(g0 - mn), p1 = __expf(g1 - mn);
        l2[j] = l2[j] * corr + rsum16(p0 + p1);
#pragma unroll
        for (int c = 0; c < 8; ++c) O2[c][j] *= corr;
        p2w[(j + half8) * 32 + ln15]      = (bf16)p0;
        p2w[(j + half8) * 32 + 16 + ln15] = (bf16)p1;
      }
    }
    v16bf pA, p2A;
    {
      const bf16* pap = pw + ln15 * 32 + (hi ? 16 : 0);
      ((v8bf*)&pA)[0] = *(const v8bf*)pap;
      ((v8bf*)&pA)[1] = *(const v8bf*)(pap + 8);
      const bf16* p2ap = p2w + ln15 * 32 + (hi ? 16 : 0);
      ((v8bf*)&p2A)[0] = *(const v8bf*)p2ap;
      ((v8bf*)&p2A)[1] = *(const v8bf*)(p2ap + 8);
    }
#pragma unroll
    for (int c = 0; c < 8; ++c) {
      v16bf vB;
      const bf16* vp = vb + (size_t)(c * 16 + ln15) * TT + sb + (hi ? 16 : 0);
      ((v8bf*)&vB)[0] = *(const v8bf*)vp;
      ((v8bf*)&vB)[1] = *(const v8bf*)(vp + 8);
      O[c]  = wmma_bf16(pA,  vB, O[c]);
      O2[c] = wmma_bf16(p2A, vB, O2[c]);
    }
  }

  float invl[8], invl2[8];
#pragma unroll
  for (int j = 0; j < 8; ++j) { invl[j] = 1.f / l[j]; invl2[j] = 1.f / l2[j]; }
#pragma unroll
  for (int c = 0; c < 8; ++c) {
#pragma unroll
    for (int j = 0; j < 8; ++j) {
      float o  = O[c][j] * invl[j];
      float o2 = O2[c][j] * invl2[j];
      float f = o + hs * (o2 - o);
      int t = qbase + j + half8;
      attn_out[((size_t)(bidx * TT + t)) * CC + h * DD + c * 16 + ln15] = (bf16)f;
    }
  }
}

// ---------------------------------------------------------------------------
// Hexagram VQ block: x2 = x1 + hex_scale * (sign(LN(x1) @ toq^T) @ fromq^T)
// ---------------------------------------------------------------------------
__global__ __launch_bounds__(256) void hex_kernel(
    const float* __restrict__ x1, const float* __restrict__ g,
    const float* __restrict__ b, const float* __restrict__ toq,
    const float* __restrict__ fromq, const float* __restrict__ hexscale,
    float* __restrict__ x2) {
  int row = blockIdx.x, tid = threadIdx.x;
  const float* xr = x1 + (size_t)row * CC;
  float xv[4], s = 0.f, ss = 0.f;
#pragma unroll
  for (int it = 0; it < 4; ++it) {
    float v = xr[tid + 256 * it];
    xv[it] = v; s += v; ss += v * v;
  }
  for (int m = 16; m >= 1; m >>= 1) { s += __shfl_xor(s, m, 32); ss += __shfl_xor(ss, m, 32); }
  __shared__ float shs[8], shss[8], stats[2], zacc[8];
  if ((tid & 31) == 0) { shs[tid >> 5] = s; shss[tid >> 5] = ss; }
  if (tid < 8) zacc[tid] = 0.f;
  __syncthreads();
  if (tid == 0) {
    float S = 0.f, SS = 0.f;
    for (int i = 0; i < 8; ++i) { S += shs[i]; SS += shss[i]; }
    float mean = S / CC, var = SS / CC - mean * mean;
    stats[0] = mean; stats[1] = rsqrtf(var + 1e-5f);
  }
  __syncthreads();
  float mean = stats[0], rstd = stats[1];
  float acc[6] = {0.f, 0.f, 0.f, 0.f, 0.f, 0.f};
#pragma unroll
  for (int it = 0; it < 4; ++it) {
    int c = tid + 256 * it;
    float hcv = (xv[it] - mean) * rstd * g[c] + b[c];
#pragma unroll
    for (int j = 0; j < 6; ++j) acc[j] += hcv * toq[j * CC + c];
  }
  for (int m = 16; m >= 1; m >>= 1)
#pragma unroll
    for (int j = 0; j < 6; ++j) acc[j] += __shfl_xor(acc[j], m, 32);
  if ((tid & 31) == 0)
    for (int j = 0; j < 6; ++j) atomicAdd(&zacc[j], acc[j]);
  __syncthreads();
  if (tid < 6) zacc[tid] = (zacc[tid] >= 0.f) ? 1.f : -1.f;
  __syncthreads();
  float hscl = hexscale[0];
#pragma unroll
  for (int it = 0; it < 4; ++it) {
    int c = tid + 256 * it;
    float d = 0.f;
#pragma unroll
    for (int j = 0; j < 6; ++j) d += zacc[j] * fromq[c * 6 + j];
    x2[(size_t)row * CC + c] = xv[it] + hscl * d;
  }
}

// ---------------------------------------------------------------------------
// Host-side orchestration
// ---------------------------------------------------------------------------
extern "C" void kernel_launch(void* const* d_in, const int* in_sizes, int n_in,
                              void* d_out, int out_size, void* d_ws, size_t ws_size,
                              hipStream_t stream) {
  (void)in_sizes; (void)n_in; (void)out_size; (void)ws_size;
  const float* x          = (const float*)d_in[0];
  const float* ln_attn_g  = (const float*)d_in[1];
  const float* ln_attn_b  = (const float*)d_in[2];
  const float* qkv_w      = (const float*)d_in[3];
  const float* out_w      = (const float*)d_in[4];
  const float* head_scale = (const float*)d_in[5];
  const float* ln_hex_g   = (const float*)d_in[6];
  const float* ln_hex_b   = (const float*)d_in[7];
  const float* to_qd_w    = (const float*)d_in[8];
  const float* from_qd_w  = (const float*)d_in[9];
  const float* hex_scale  = (const float*)d_in[10];
  const float* ln_ffn_g   = (const float*)d_in[11];
  const float* ln_ffn_b   = (const float*)d_in[12];
  const float* ffn_w1     = (const float*)d_in[13];
  const float* ffn_b1     = (const float*)d_in[14];
  const float* ffn_w2     = (const float*)d_in[15];
  const float* ffn_b2     = (const float*)d_in[16];
  float* out = (float*)d_out;

  char* ws = (char*)d_ws;
  const size_t MBy = 1024ull * 1024ull;
  bf16*  Wq   = (bf16*)(ws + 0);            // 6 MB   (3C x C)
  bf16*  Wo   = (bf16*)(ws + 6 * MBy);      // 2 MB   (C x C)
  bf16*  W1   = (bf16*)(ws + 8 * MBy);      // 8 MB   (FFN x C)
  bf16*  W2   = (bf16*)(ws + 16 * MBy);     // 8 MB   (C x FFN)
  bf16*  Hb   = (bf16*)(ws + 24 * MBy);     // 8 MB   LN output, reused 2x
  float* Qkv  = (float*)(ws + 32 * MBy);    // 48 MB  qkv f32 (dead after rope)
  bf16*  Mid  = (bf16*)(ws + 32 * MBy);     // 32 MB  FFN mid (reuses qkv region)
  bf16*  Qb   = (bf16*)(ws + 80 * MBy);     // 8 MB
  bf16*  Kb   = (bf16*)(ws + 88 * MBy);     // 8 MB
  bf16*  Vt   = (bf16*)(ws + 96 * MBy);     // 8 MB
  float* Qp   = (float*)(ws + 104 * MBy);   // 128 KB
  float* Kp   = (float*)(ws + 104 * MBy + 256 * 1024);
  bf16*  Attn = (bf16*)(ws + 105 * MBy);    // 8 MB
  float* X1   = (float*)(ws + 113 * MBy);   // 16 MB
  float* X2   = (float*)(ws + 129 * MBy);   // 16 MB  (total 145 MB)

  dim3 blk(256);

  cvt_bf16_kernel<<<(3 * CC * CC) / 256, blk, 0, stream>>>(qkv_w, Wq, 3 * CC * CC);
  cvt_bf16_kernel<<<(CC * CC) / 256, blk, 0, stream>>>(out_w, Wo, CC * CC);
  cvt_bf16_kernel<<<(FFN_ * CC) / 256, blk, 0, stream>>>(ffn_w1, W1, FFN_ * CC);
  cvt_bf16_kernel<<<(CC * FFN_) / 256, blk, 0, stream>>>(ffn_w2, W2, CC * FFN_);

  ln_bf16_kernel<<<ROWS, blk, 0, stream>>>(x, ln_attn_g, ln_attn_b, Hb);

  gemm_bf16_wmma<0><<<dim3(3 * CC / 64, ROWS / 128), blk, 0, stream>>>(
      Hb, Wq, ROWS, 3 * CC, CC, Qkv, nullptr, nullptr, nullptr);

  rope_split_kernel<<<(BB * HH * TT * 64) / 256, blk, 0, stream>>>(Qkv, Qb, Kb, Vt);
  qpkp_kernel<<<(BB * HH * TT) / 256, blk, 0, stream>>>(Qkv, Qp, Kp);

  flash_kernel<<<(BB * HH * (TT / 16)) / 8, blk, 0, stream>>>(
      Qb, Kb, Vt, Qp, Kp, head_scale, Attn);

  gemm_bf16_wmma<2><<<dim3(CC / 64, ROWS / 128), blk, 0, stream>>>(
      Attn, Wo, ROWS, CC, CC, X1, nullptr, nullptr, x);

  hex_kernel<<<ROWS, blk, 0, stream>>>(X1, ln_hex_g, ln_hex_b, to_qd_w, from_qd_w,
                                       hex_scale, X2);

  ln_bf16_kernel<<<ROWS, blk, 0, stream>>>(X2, ln_ffn_g, ln_ffn_b, Hb);
  gemm_bf16_wmma<1><<<dim3(FFN_ / 64, ROWS / 128), blk, 0, stream>>>(
      Hb, W1, ROWS, FFN_, CC, nullptr, Mid, ffn_b1, nullptr);
  gemm_bf16_wmma<3><<<dim3(CC / 64, ROWS / 128), blk, 0, stream>>>(
      Mid, W2, ROWS, CC, FFN_, out, nullptr, ffn_b2, X2);
}
